// LMHSA_73967926771816
// MI455X (gfx1250) — compile-verified, matrix operand order checked by reading, not run on hardware
//
#include <hip/hip_runtime.h>
#include <hip/hip_bf16.h>

// ---------------------------------------------------------------------------
// LMHSA for MI455X (gfx1250, wave32). Memory-bound op (~150MB traffic vs
// ~11 GFLOP) -> f16 intermediates + v_wmma_f32_16x16x32_f16 everywhere,
// f32 softmax/LayerNorm/bias/residual. Attention uses online (flash-style)
// softmax over two 128-key chunks so only 8 S-tiles are live at once,
// keeping the wave under 256 VGPRs (no scratch spills).
// ---------------------------------------------------------------------------

#define B_SZ   32
#define D_CH   64
#define HW     64
#define STRIDE 4
#define HEADS  2
#define DK     32
#define DV     32
#define N_TOK  4096
#define M_TOK  256
#define EPS_LN 1e-5f

typedef _Float16 v16h __attribute__((ext_vector_type(16)));
typedef float    v8f  __attribute__((ext_vector_type(8)));
typedef _Float16 v2h  __attribute__((ext_vector_type(2)));

// K-index of the j-th packed f16 pair in a 16-bit A/B WMMA fragment
// (CDNA5 ISA 7.12.2: VGPR0..3 -> K {0..7}+8*half, VGPR4..7 -> K {16..23}+8*half)
__device__ __forceinline__ int kmap(int j, int half) {
  return ((j & 4) << 2) + 8 * half + 2 * (j & 3);
}

// Load a 16x32 (or 32x16) f16 fragment given this lane's row/col base pointer.
// All pair offsets are even -> 32-bit aligned loads.
__device__ __forceinline__ v16h load_frag16(const _Float16* base, int half) {
  v16h f;
#pragma unroll
  for (int j = 0; j < 8; ++j) {
    v2h p = *(const v2h*)(base + kmap(j, half));
    f[2 * j]     = p.x;
    f[2 * j + 1] = p.y;
  }
  return f;
}

// ---------------------------------------------------------------------------
// Kernel 1: per-batch LayerNorm statistics (mean, rstd over C*H*W)
// ---------------------------------------------------------------------------
__global__ void lmhsa_ln_stats(const float* __restrict__ x,
                               float* __restrict__ stats) {
  int b = blockIdx.x;
  const float* xb = x + (size_t)b * D_CH * N_TOK;
  float s1 = 0.f, s2 = 0.f;
  for (int i = threadIdx.x; i < D_CH * N_TOK; i += 256) {
    float v = xb[i];
    s1 += v;
    s2 += v * v;
  }
  __shared__ float r1[256], r2[256];
  r1[threadIdx.x] = s1;
  r2[threadIdx.x] = s2;
  __syncthreads();
  for (int off = 128; off > 0; off >>= 1) {
    if (threadIdx.x < off) {
      r1[threadIdx.x] += r1[threadIdx.x + off];
      r2[threadIdx.x] += r2[threadIdx.x + off];
    }
    __syncthreads();
  }
  if (threadIdx.x == 0) {
    float inv = 1.0f / (float)(D_CH * N_TOK);
    float mu  = r1[0] * inv;
    float var = r2[0] * inv - mu * mu;
    stats[b]        = mu;
    stats[B_SZ + b] = rsqrtf(var + EPS_LN);
  }
}

// ---------------------------------------------------------------------------
// Kernel 2: normalize + transpose: xn[b][token][c] = (x[b][c][token]-mu)*rstd
// ---------------------------------------------------------------------------
__global__ void lmhsa_normalize(const float* __restrict__ x,
                                const float* __restrict__ stats,
                                _Float16* __restrict__ xn) {
  size_t idx = (size_t)blockIdx.x * 256 + threadIdx.x;   // b*c*token
  int n = (int)(idx & (N_TOK - 1));
  int c = (int)((idx >> 12) & (D_CH - 1));
  int b = (int)(idx >> 18);
  float mu = stats[b], rs = stats[B_SZ + b];
  xn[((size_t)(b * N_TOK + n)) * D_CH + c] = (_Float16)((x[idx] - mu) * rs);
}

// ---------------------------------------------------------------------------
// Kernel 3: depthwise 4x4 stride-4 conv on raw x -> kv_in[b][m][c] (f16)
// ---------------------------------------------------------------------------
__global__ void lmhsa_dwconv(const float* __restrict__ x,
                             const float* __restrict__ Wdw,
                             const float* __restrict__ bdw,
                             _Float16* __restrict__ kv) {
  int idx = blockIdx.x * 256 + threadIdx.x;   // b*m*c = 32*256*64
  int c = idx & 63;
  int m = (idx >> 6) & 255;
  int b = idx >> 14;
  int kh = m >> 4, kw = m & 15;
  const float* xp = x + (((size_t)(b * D_CH + c) * HW) + kh * STRIDE) * HW + kw * STRIDE;
  const float* wp = Wdw + c * 16;
  float acc = bdw[c];
#pragma unroll
  for (int i = 0; i < 4; ++i)
#pragma unroll
    for (int j = 0; j < 4; ++j)
      acc += xp[i * HW + j] * wp[i * 4 + j];
  kv[(size_t)(b * M_TOK + m) * D_CH + c] = (_Float16)acc;
}

// ---------------------------------------------------------------------------
// Kernel 4: generic 64-wide projection  out = A(tokens x 64) @ W^T + bias
// A is f16 row-major [b][token][64]; W,bias f32. FINAL=0: store f16.
// FINAL=1: raw-view remap (token -> (c,h), col -> w) + residual, f32 out.
// Block: 256 threads = 8 waves; wave w owns 16-token row block; K=64 via
// two chained v_wmma_f32_16x16x32_f16 per 16-col tile.
// ---------------------------------------------------------------------------
template <int FINAL>
__global__ void lmhsa_proj64(const _Float16* __restrict__ A,
                             const float* __restrict__ W,
                             const float* __restrict__ bias,
                             void* __restrict__ outp,
                             const float* __restrict__ xres,
                             int tokens) {
  __shared__ __align__(16) _Float16 wl[64 * 64];  // W as f16, [o][c] row-major
  __shared__ float bl[64];
  int tiles_per_b = tokens / 128;
  int b  = blockIdx.x / tiles_per_b;
  int tq = (blockIdx.x % tiles_per_b) * 128;
  int tid = threadIdx.x;
  for (int i = tid; i < 64 * 64; i += 256) wl[i] = (_Float16)W[i];
  if (tid < 64) bl[tid] = bias[tid];
  __syncthreads();

  int w = tid >> 5, lane = tid & 31, half = lane >> 4, ln = lane & 15;
  const _Float16* arow = A + ((size_t)(b * tokens + tq + w * 16 + ln)) * 64;
  v16h a0 = load_frag16(arow, half);
  v16h a1 = load_frag16(arow + 32, half);

#pragma unroll
  for (int cb = 0; cb < 4; ++cb) {
    // B[k=c][n=o] = W[o][c]  -> contiguous pairs along c in wl rows
    const _Float16* wcol = wl + (cb * 16 + ln) * 64;
    v16h b0 = load_frag16(wcol, half);
    v16h b1 = load_frag16(wcol + 32, half);
    v8f c = {};
    c = __builtin_amdgcn_wmma_f32_16x16x32_f16(false, a0, false, b0, (short)0, c, false, false);
    c = __builtin_amdgcn_wmma_f32_16x16x32_f16(false, a1, false, b1, (short)0, c, false, false);
    int o = cb * 16 + ln;
    float bv = bl[o];
#pragma unroll
    for (int r = 0; r < 8; ++r) {
      int token = tq + w * 16 + r + 8 * half;   // C-layout row
      float val = c[r] + bv;
      if (FINAL) {
        int cch = token >> 6, hh = token & 63;  // raw .view(b,c,h,w) remap
        size_t oi = (((size_t)(b * 64 + cch) * 64 + hh) * 64 + o);
        ((float*)outp)[oi] = val + xres[oi];
      } else {
        ((_Float16*)outp)[((size_t)(b * tokens + token)) * 64 + o] = (_Float16)val;
      }
    }
  }
}

// ---------------------------------------------------------------------------
// Kernel 5: attention with online softmax. grid = B*HEADS*(N/128),
// block = 256 (8 waves). Each wave owns 16 queries. Keys processed in two
// 128-key chunks: per chunk 8 S-tile WMMAs (only 64 live acc VGPRs),
// scale + additive bias, running-max/sum softmax update with output
// rescaling, then P re-swizzled through a 1KB per-wave LDS bounce buffer
// and multiplied against V^T staged in LDS (8 WMMAs per chunk).
// ---------------------------------------------------------------------------
__global__ void __launch_bounds__(256, 1)
lmhsa_attn(const _Float16* __restrict__ q,
           const _Float16* __restrict__ kmat,
           const _Float16* __restrict__ vmat,
           const float* __restrict__ bias,
           _Float16* __restrict__ tmp) {
  __shared__ __align__(16) _Float16 ks[M_TOK * DK];      // [key][dk]   16KB
  __shared__ __align__(16) _Float16 vsT[DV * M_TOK];     // [dv][key]   16KB
  __shared__ __align__(16) _Float16 pbuf[8][16 * 32];    // per-wave P   8KB
  int qt   = blockIdx.x & 31;
  int head = (blockIdx.x >> 5) & 1;
  int b    = blockIdx.x >> 6;
  int tq   = qt * 128;
  int tid  = threadIdx.x;

  {  // stage K (row-major) and V (transposed), one key row per thread
    int m = tid;
    const _Float16* ksrc = kmat + ((size_t)(b * M_TOK + m)) * 64 + head * DK;
    uint4* dstv = (uint4*)&ks[m * DK];
    const uint4* srcv = (const uint4*)ksrc;
    dstv[0] = srcv[0]; dstv[1] = srcv[1]; dstv[2] = srcv[2]; dstv[3] = srcv[3];
    const _Float16* vsrc = vmat + ((size_t)(b * M_TOK + m)) * 64 + head * DV;
#pragma unroll
    for (int dv = 0; dv < DV; ++dv) vsT[dv * M_TOK + m] = vsrc[dv];
  }
  __syncthreads();

  int w = tid >> 5, lane = tid & 31, half = lane >> 4, ln = lane & 15;
  const _Float16* qrow =
      q + ((size_t)(b * N_TOK + tq + w * 16 + ln)) * 64 + head * DK;
  v16h qa = load_frag16(qrow, half);   // A: 16 queries x 32 dk

  const float scale = 0.17677669529663687f;  // 32^-0.5
  v8f o0 = {}, o1 = {};
  float mrun[8], lrun[8];
#pragma unroll
  for (int r = 0; r < 8; ++r) { mrun[r] = -1e30f; lrun[r] = 0.f; }

  _Float16* pb = &pbuf[w][0];

#pragma unroll
  for (int ch = 0; ch < 2; ++ch) {           // two 128-key chunks
    v8f s[8];
#pragma unroll
    for (int t = 0; t < 8; ++t) {
      int kt = ch * 8 + t;
      v16h kb = load_frag16(ks + (kt * 16 + ln) * DK, half);  // B[k=dk][n=key]
      v8f z = {};
      s[t] = __builtin_amdgcn_wmma_f32_16x16x32_f16(false, qa, false, kb,
                                                    (short)0, z, false, false);
    }
    // scale + additive relative bias
#pragma unroll
    for (int r = 0; r < 8; ++r) {
      int token = tq + w * 16 + r + 8 * half;
      const float* brow =
          bias + ((size_t)(head * N_TOK + token)) * M_TOK + ch * 128 + ln;
#pragma unroll
      for (int t = 0; t < 8; ++t)
        s[t][r] = s[t][r] * scale + brow[t * 16];
    }
    // online softmax update (per row, reduced across the 16-lane group)
#pragma unroll
    for (int r = 0; r < 8; ++r) {
      float mc = s[0][r];
#pragma unroll
      for (int t = 1; t < 8; ++t) mc = fmaxf(mc, s[t][r]);
      mc = fmaxf(mc, __shfl_xor(mc, 1, 32));
      mc = fmaxf(mc, __shfl_xor(mc, 2, 32));
      mc = fmaxf(mc, __shfl_xor(mc, 4, 32));
      mc = fmaxf(mc, __shfl_xor(mc, 8, 32));
      float mnew  = fmaxf(mrun[r], mc);
      float alpha = __expf(mrun[r] - mnew);   // 0 on first chunk
      float sum = 0.f;
#pragma unroll
      for (int t = 0; t < 8; ++t) {
        float e = __expf(s[t][r] - mnew);
        s[t][r] = e;
        sum += e;
      }
      sum += __shfl_xor(sum, 1, 32);
      sum += __shfl_xor(sum, 2, 32);
      sum += __shfl_xor(sum, 4, 32);
      sum += __shfl_xor(sum, 8, 32);
      lrun[r] = lrun[r] * alpha + sum;
      mrun[r] = mnew;
      o0[r] *= alpha;
      o1[r] *= alpha;
    }
    // P @ V for this chunk: 4 sub-chunks of 32 keys; C-layout -> A-layout
    // via the per-wave LDS bounce buffer (same-wave DScnt ordering).
#pragma unroll
    for (int kk = 0; kk < 4; ++kk) {
#pragma unroll
      for (int t2 = 0; t2 < 2; ++t2) {
        int t = kk * 2 + t2;
#pragma unroll
        for (int r = 0; r < 8; ++r) {
          int m = r + 8 * half;                          // C-layout row
          pb[m * 32 + t2 * 16 + ln] = (_Float16)s[t][r];
        }
      }
      int kbase = ch * 128 + kk * 32;
      v16h pa = load_frag16(pb + ln * 32, half);                     // A: P 16x32
      v16h b0 = load_frag16(vsT + (size_t)ln * M_TOK + kbase, half); // B: V^T dv 0-15
      v16h b1 = load_frag16(vsT + (size_t)(16 + ln) * M_TOK + kbase, half);
      o0 = __builtin_amdgcn_wmma_f32_16x16x32_f16(false, pa, false, b0,
                                                  (short)0, o0, false, false);
      o1 = __builtin_amdgcn_wmma_f32_16x16x32_f16(false, pa, false, b1,
                                                  (short)0, o1, false, false);
    }
  }

#pragma unroll
  for (int r = 0; r < 8; ++r) {
    float inv = 1.0f / lrun[r];
    int token = tq + w * 16 + r + 8 * half;
    _Float16* dst = tmp + ((size_t)(b * N_TOK + token)) * 64 + head * DV;
    dst[ln]      = (_Float16)(o0[r] * inv);
    dst[16 + ln] = (_Float16)(o1[r] * inv);
  }
}

// ---------------------------------------------------------------------------
// Host launcher
// ---------------------------------------------------------------------------
extern "C" void kernel_launch(void* const* d_in, const int* in_sizes, int n_in,
                              void* d_out, int out_size, void* d_ws, size_t ws_size,
                              hipStream_t stream) {
  (void)in_sizes; (void)n_in; (void)out_size; (void)ws_size;
  const float* x   = (const float*)d_in[0];
  const float* Wdw = (const float*)d_in[1];
  const float* bdw = (const float*)d_in[2];
  const float* Wq  = (const float*)d_in[3];
  const float* bq  = (const float*)d_in[4];
  const float* Wk  = (const float*)d_in[5];
  const float* bk  = (const float*)d_in[6];
  const float* Wv  = (const float*)d_in[7];
  const float* bv  = (const float*)d_in[8];
  const float* Wo  = (const float*)d_in[9];
  const float* bo  = (const float*)d_in[10];
  const float* Bb  = (const float*)d_in[11];
  float* out = (float*)d_out;

  char* ws = (char*)d_ws;
  size_t off = 0;
  float* stats  = (float*)(ws + off); off += 1024;
  _Float16* xn  = (_Float16*)(ws + off); off += (size_t)B_SZ * N_TOK * 64 * 2;  // 16MB
  _Float16* qm  = (_Float16*)(ws + off); off += (size_t)B_SZ * N_TOK * 64 * 2;  // 16MB
  _Float16* kv  = (_Float16*)(ws + off); off += (size_t)B_SZ * M_TOK * 64 * 2;  //  1MB
  _Float16* km  = (_Float16*)(ws + off); off += (size_t)B_SZ * M_TOK * 64 * 2;  //  1MB
  _Float16* vm  = (_Float16*)(ws + off); off += (size_t)B_SZ * M_TOK * 64 * 2;  //  1MB
  _Float16* tmp = (_Float16*)(ws + off); off += (size_t)B_SZ * N_TOK * 64 * 2;  // 16MB

  lmhsa_ln_stats <<<B_SZ, 256, 0, stream>>>(x, stats);
  lmhsa_normalize<<<(B_SZ * D_CH * N_TOK) / 256, 256, 0, stream>>>(x, stats, xn);
  lmhsa_dwconv   <<<(B_SZ * M_TOK * D_CH) / 256, 256, 0, stream>>>(x, Wdw, bdw, kv);
  lmhsa_proj64<0><<<B_SZ * (N_TOK / 128), 256, 0, stream>>>(xn, Wq, bq, (void*)qm, nullptr, N_TOK);
  lmhsa_proj64<0><<<B_SZ * (M_TOK / 128), 256, 0, stream>>>(kv, Wk, bk, (void*)km, nullptr, M_TOK);
  lmhsa_proj64<0><<<B_SZ * (M_TOK / 128), 256, 0, stream>>>(kv, Wv, bv, (void*)vm, nullptr, M_TOK);
  lmhsa_attn     <<<B_SZ * HEADS * (N_TOK / 128), 256, 0, stream>>>(qm, km, vm, Bb, tmp);
  lmhsa_proj64<1><<<B_SZ * (N_TOK / 128), 256, 0, stream>>>(tmp, Wo, bo, (void*)out, x, N_TOK);
}
